// Fusionloss_45552423141545
// MI455X (gfx1250) — matrix operand order repaired and checked by reading, not compiled, not run on hardware
//
#include <hip/hip_runtime.h>
#include <math.h>

typedef __attribute__((ext_vector_type(16))) _Float16 v16h;
typedef __attribute__((ext_vector_type(8)))  float    v8f;

#define IMG_HW   512
#define NPIX_F   8388608.0f      // 32*512*512
#define WIN      11
#define C2_CONST 0.0009f         // 0.03^2

// per-wave tile geometry
#define TROWS 32                 // output rows per wave strip
#define TCOLS 16                 // output cols per wave strip
#define XROWS 48                 // staged rows: row0-5 .. row0+42
#define XCOLS 40                 // staged cols: col0-8 .. col0+31 (16B-aligned halo)
#define CHALO 8                  // left halo in staged window
#define HROWP 56                 // H buffer row length in halfs (48 rows + pad)

#define X_BYTES   (3 * XROWS * XCOLS * 4)   // 23040
#define H_BYTES   (8 * 16 * HROWP * 2)      // 14336
#define WAVE_BYTES (X_BYTES + H_BYTES)      // 37376
#define WAVES_PER_BLOCK 4
#define LDS_BYTES (WAVES_PER_BLOCK * WAVE_BYTES)  // 149504

__global__ void __launch_bounds__(128)
fusion_main(const float* __restrict__ gen, const float* __restrict__ ir,
            const float* __restrict__ vis, float* __restrict__ ws)
{
    extern __shared__ float4 smem_f4[];
    char* smem = (char*)smem_f4;

    const int lane = threadIdx.x & 31;
    const int wave = threadIdx.x >> 5;
    const int hi   = lane >> 4;      // half-wave select
    const int ln   = lane & 15;

    char*     wb = smem + wave * WAVE_BYTES;
    float*    Xs = (float*)wb;                      // [3][XROWS][XCOLS] f32
    _Float16* Hs = (_Float16*)(wb + X_BYTES);       // [8][16][HROWP]   f16

    // ---- tile decode: 32 batches * 16 row-strips * 32 col-tiles = 16384 waves
    const int tile = blockIdx.x * WAVES_PER_BLOCK + wave;
    const int b    = tile >> 9;              // / 512
    const int rem  = tile & 511;
    const int row0 = (rem >> 5) * TROWS;
    const int col0 = (rem & 31) * TCOLS;
    const size_t plane = (size_t)b * IMG_HW * IMG_HW;
    const float* srcs[3] = { gen + plane, ir + plane, vis + plane }; // x1,x2,x3

    // ---- stage zero-padded windows of all three images into LDS
    const bool interior = (row0 >= 32) && (row0 <= 448) &&
                          (col0 >= 16) && (col0 <= 480);
    if (interior) {
        // direct HBM -> LDS async copies (16B chunks, all aligned, all in-bounds)
        // 3 images x 48 rows x 160B = 45 b128 instructions per wave
        const unsigned xbase = (unsigned)(uintptr_t)Xs;
        #pragma unroll
        for (int img = 0; img < 3; ++img) {
            const char* gbase = (const char*)srcs[img] +
                                ((size_t)(row0 - 5) * IMG_HW + (col0 - CHALO)) * 4;
            const unsigned lbase = xbase + img * XROWS * XCOLS * 4;
            #pragma unroll
            for (int it = 0; it < 15; ++it) {
                int q   = it * 32 + lane;        // chunk id 0..479
                int r   = q / 10;                // staged row
                int cc  = q - r * 10;            // 16B chunk within row
                unsigned ldsoff = lbase + r * (XCOLS * 4) + cc * 16;
                unsigned long long ga =
                    (unsigned long long)(uintptr_t)(gbase + (size_t)r * (IMG_HW * 4) + cc * 16);
                asm volatile("global_load_async_to_lds_b128 %0, %1, off"
                             :: "v"(ldsoff), "v"(ga) : "memory");
            }
        }
        asm volatile("s_wait_asynccnt 0" ::: "memory");
    } else {
        // boundary tiles: guarded loads with zero padding (matches SAME conv padding)
        #pragma unroll
        for (int img = 0; img < 3; ++img) {
            const float* src = srcs[img];
            float* xd = Xs + img * XROWS * XCOLS;
            for (int r = 0; r < XROWS; ++r) {
                int gr = row0 - 5 + r;
                #pragma unroll
                for (int base = 0; base < XCOLS; base += 32) {
                    int cc = base + lane;
                    if (cc < XCOLS) {
                        int gc = col0 - CHALO + cc;
                        float v = 0.f;
                        if ((unsigned)gr < 512u && (unsigned)gc < 512u)
                            v = src[gr * IMG_HW + gc];
                        xd[r * XCOLS + cc] = v;
                    }
                }
            }
        }
    }

    // ---- 1D gaussian (sigma=1.5, 11 taps, normalized)
    float g[WIN];
    {
        float s = 0.f;
        #pragma unroll
        for (int i = 0; i < WIN; ++i) {
            float x = (float)(i - 5);
            g[i] = __expf(-(x * x) * (1.0f / (2.0f * 1.5f * 1.5f)));
            s += g[i];
        }
        float inv = 1.f / s;
        #pragma unroll
        for (int i = 0; i < WIN; ++i) g[i] *= inv;
    }

    // ---- constant band-matrix WMMA operands
    // Sh: B-layout (32x16), S[k][n] = g[k-n-3]   (horizontal, window starts at col0-8)
    // Wh: A-layout (16x32), W[m][k] = g[k-m]     (vertical,   window starts at row0-5)
    v16h Sh, Wh;
    #pragma unroll
    for (int j = 0; j < 16; ++j) {
        int kB = (j & 7) + 8 * hi + ((j & 8) ? 16 : 0);  // B: K table order, N = ln
        int dB = kB - ln - 3;
        Sh[j] = (dB >= 0 && dB <= 10) ? (_Float16)g[dB] : (_Float16)0.f;
        int kA = j + 16 * hi;                       // A: K = j + 16*hi, M = ln
        int dA = kA - ln;
        Wh[j] = (dA >= 0 && dA <= 10) ? (_Float16)g[dA] : (_Float16)0.f;
    }

    const v8f zc = {};

    // ---- horizontal gaussian pass: 3 row-blocks x 8 fields, one WMMA each
    #pragma unroll
    for (int rb = 0; rb < 3; ++rb) {
        const int row = rb * 16 + ln;           // A-layout: M = ln for both halves
        v16h a[3];
        #pragma unroll
        for (int img = 0; img < 3; ++img) {
            const float* xr = Xs + (img * XROWS + row) * XCOLS;
            float4 f0 = *(const float4*)(xr + 8 * hi);
            float4 f1 = *(const float4*)(xr + 8 * hi + 4);
            float4 f2 = *(const float4*)(xr + 16 + 8 * hi);
            float4 f3 = *(const float4*)(xr + 20 + 8 * hi);
            v16h t;
            t[0]=(_Float16)f0.x; t[1]=(_Float16)f0.y; t[2]=(_Float16)f0.z; t[3]=(_Float16)f0.w;
            t[4]=(_Float16)f1.x; t[5]=(_Float16)f1.y; t[6]=(_Float16)f1.z; t[7]=(_Float16)f1.w;
            t[8]=(_Float16)f2.x; t[9]=(_Float16)f2.y; t[10]=(_Float16)f2.z; t[11]=(_Float16)f2.w;
            t[12]=(_Float16)f3.x; t[13]=(_Float16)f3.y; t[14]=(_Float16)f3.z; t[15]=(_Float16)f3.w;
            a[img] = t;
        }
        v16h ops[8] = { a[0], a[1], a[2],
                        a[0]*a[0], a[1]*a[1], a[2]*a[2],
                        a[0]*a[1], a[0]*a[2] };
        #pragma unroll
        for (int f = 0; f < 8; ++f) {
            v8f c = __builtin_amdgcn_wmma_f32_16x16x32_f16(
                        false, ops[f], false, Sh, (short)0, zc, false, false);
            // C layout: VGPR r -> M = r + 8*hi, N = ln.  Store column-major H (16B chunk).
            _Float16* hp = Hs + (f * 16 + ln) * HROWP + rb * 16 + 8 * hi;
            union { _Float16 h[8]; uint4 u; } tt;
            #pragma unroll
            for (int r = 0; r < 8; ++r) tt.h[r] = (_Float16)c[r];
            *(uint4*)hp = tt.u;
        }
    }

    // ---- vertical gaussian pass + SSIM map accumulation
    float s_ssim = 0.f;
    #pragma unroll
    for (int vb = 0; vb < 2; ++vb) {
        v8f V[8];
        #pragma unroll
        for (int f = 0; f < 8; ++f) {
            // B layout: K = j + 16*hi, N = ln  ->  H rows [16*vb + 16*hi + j]
            const _Float16* hp = Hs + (f * 16 + ln) * HROWP + vb * 16 + 16 * hi;
            union { v16h v; uint4 u[2]; } tb;
            tb.u[0] = *(const uint4*)hp;
            tb.u[1] = *(const uint4*)(hp + 8);
            V[f] = __builtin_amdgcn_wmma_f32_16x16x32_f16(
                       false, Wh, false, tb.v, (short)0, zc, false, false);
        }
        #pragma unroll
        for (int r = 0; r < 8; ++r) {
            float mu1 = V[0][r], mu2 = V[1][r], mu3 = V[2][r];
            float s1  = V[3][r] - mu1 * mu1;
            float s2  = V[4][r] - mu2 * mu2;
            float s3  = V[5][r] - mu3 * mu3;
            float c12 = V[6][r] - mu1 * mu2;
            float c13 = V[7][r] - mu1 * mu3;
            float x2  = sqrtf(s2);
            float x3  = sqrtf(s3);
            float m12 = (2.f * c12 + C2_CONST) / (s1 + s2 + C2_CONST);
            float m13 = (2.f * c13 + C2_CONST) / (s1 + s3 + C2_CONST);
            s_ssim += (fabsf(x2) >= fabsf(x3)) ? m12 : m13;  // NaN -> m13, matches jnp.where
        }
    }

    // ---- sobel gradient loss + intensity loss (VALU, from staged LDS)
    float s_in = 0.f, s_grad = 0.f;
    {
        const int c = ln + CHALO;               // window col of output pixel
        for (int k = 0; k < 16; ++k) {
            int r = hi + 2 * k + 5;             // window row of output pixel
            float m[3], ctr[3];
            #pragma unroll
            for (int img = 0; img < 3; ++img) {
                const float* x0 = Xs + img * XROWS * XCOLS + (r - 1) * XCOLS + c;
                const float* x1 = x0 + XCOLS;
                const float* x2 = x1 + XCOLS;
                float a00 = x0[-1], a01 = x0[0], a02 = x0[1];
                float a10 = x1[-1],              a12 = x1[1];
                float a20 = x2[-1], a21 = x2[0], a22 = x2[1];
                float gx = (a02 - a00) + 2.f * (a12 - a10) + (a22 - a20);
                float gy = (a00 + 2.f * a01 + a02) - (a20 + 2.f * a21 + a22);
                m[img]   = fabsf(gx) + fabsf(gy);
                ctr[img] = x1[0];
            }
            s_in   += fabsf(ctr[0] - fmaxf(ctr[1], ctr[2]));   // |gen - max(ir,vis)|
            s_grad += fabsf(m[0]  - fmaxf(m[1],  m[2]));       // |S(gen) - max(S(ir),S(vis))|
        }
    }

    // ---- wave32 reduction, one atomic per wave
    #pragma unroll
    for (int off = 16; off > 0; off >>= 1) {
        s_in   += __shfl_xor(s_in,   off, 32);
        s_grad += __shfl_xor(s_grad, off, 32);
        s_ssim += __shfl_xor(s_ssim, off, 32);
    }
    if (lane == 0) {
        atomicAdd(ws + 0, s_in);
        atomicAdd(ws + 1, s_grad);
        atomicAdd(ws + 2, s_ssim);
    }
}

__global__ void fusion_finalize(const float* __restrict__ ws, float* __restrict__ out)
{
    const float inv = 1.0f / NPIX_F;
    out[0] = 1.5f * ws[0] * inv;
    float ssim_mean = ws[2] * inv;
    out[1] = 0.5f * (1.0f - ssim_mean) + ws[1] * inv;
}

extern "C" void kernel_launch(void* const* d_in, const int* in_sizes, int n_in,
                              void* d_out, int out_size, void* d_ws, size_t ws_size,
                              hipStream_t stream) {
    (void)in_sizes; (void)n_in; (void)out_size; (void)ws_size;
    const float* vis = (const float*)d_in[0];
    const float* ir  = (const float*)d_in[1];
    const float* gen = (const float*)d_in[2];
    float* ws  = (float*)d_ws;
    float* out = (float*)d_out;

    hipMemsetAsync(ws, 0, 3 * sizeof(float), stream);

    // 16384 wave-tiles / 4 waves per block
    dim3 grid(4096), block(128);
    fusion_main<<<grid, block, LDS_BYTES, stream>>>(gen, ir, vis, ws);
    fusion_finalize<<<1, 1, 0, stream>>>(ws, out);
}
